// GCN_82205674046056
// MI455X (gfx1250) — matrix operand order repaired and checked by reading, not compiled, not run on hardware
//
#include <hip/hip_runtime.h>
#include <hip/hip_bf16.h>

// ---------------------------------------------------------------------------
// GCN forward for MI455X (gfx1250, wave32).
//   x[N,256] @ W1[256,128] -> gather/scatter over 3.2M edges -> +b1, relu
//   -> @ W2[128,2] -> gather/scatter -> +b2 -> mean-pool by graph -> log_softmax
// GEMM1 uses v_wmma_f32_16x16x32_bf16 (f32 accumulate). Aggregation uses
// wave-per-edge float4 gathers + global f32 atomics (L2-resident: h1 = 51MB,
// L2 = 192MB).
// ---------------------------------------------------------------------------

namespace {
constexpr int   NN  = 100000;   // nodes
constexpr int   EE  = 3200000;  // directed edges
constexpr int   FIN = 256;      // input features
constexpr int   HH  = 128;      // hidden dim
constexpr int   CC  = 2;        // classes
constexpr int   GG  = 128;      // graphs
constexpr int   MTILES = NN / 16;  // 6250, exact
}

typedef __attribute__((ext_vector_type(16))) __bf16 v16bf;
typedef __attribute__((ext_vector_type(8)))  __bf16 v8bf;
typedef __attribute__((ext_vector_type(8)))  float  v8f;

static __device__ __forceinline__ __bf16 to_bf16(float f) {
  unsigned u = __builtin_bit_cast(unsigned, f);
  u += 0x7FFFu + ((u >> 16) & 1u);          // round-to-nearest-even
  unsigned short h = (unsigned short)(u >> 16);
  return __builtin_bit_cast(__bf16, h);
}

// ---------------------------------------------------------------------------
// 0) init: zero accumulation buffers, deg = 1.0 (self loop)
// ---------------------------------------------------------------------------
__global__ void init_ws(float* __restrict__ ag1, float* __restrict__ ag2,
                        float* __restrict__ deg, float* __restrict__ gsum,
                        float* __restrict__ gcnt) {
  long i = (long)blockIdx.x * blockDim.x + threadIdx.x;
  long tot = (long)NN * HH;
  if (i < tot)          ag1[i] = 0.0f;
  if (i < (long)NN * CC) ag2[i] = 0.0f;
  if (i < NN)           deg[i] = 1.0f;     // A + I: self loop contributes 1
  if (i < GG * CC)      gsum[i] = 0.0f;
  if (i < GG)           gcnt[i] = 0.0f;
}

// ---------------------------------------------------------------------------
// 1) in-degree over dst (plus the self loop already in init), then deg^-1/2
// ---------------------------------------------------------------------------
__global__ void deg_kernel(const int* __restrict__ dst, float* __restrict__ deg) {
  long e = (long)blockIdx.x * blockDim.x + threadIdx.x;
  if (e < EE) atomicAdd(&deg[dst[e]], 1.0f);
}

__global__ void dis_kernel(const float* __restrict__ deg, float* __restrict__ dis) {
  long n = (long)blockIdx.x * blockDim.x + threadIdx.x;
  if (n < NN) {
    float d = deg[n];
    dis[n] = d > 0.0f ? rsqrtf(d) : 0.0f;
  }
}

// ---------------------------------------------------------------------------
// 2) h1 = x @ W1  via WMMA bf16 -> f32.
//    Block: 256 threads = 8 waves. W1 staged once per block in 64KB LDS (bf16,
//    XOR-swizzled 32B chunks for bank-conflict-free B-fragment reads).
//    Each wave: one 16-row M tile x full 128 cols (8 accumulators of v8f).
// ---------------------------------------------------------------------------
__global__ __launch_bounds__(256) void gemm1_wmma(const float* __restrict__ x,
                                                  const float* __restrict__ W1,
                                                  float* __restrict__ h1) {
  __shared__ __bf16 w1s[FIN * HH];   // exactly 64 KB

  const int tid = threadIdx.x;
  // Cooperative fill: W1 is row-major [k][n]; swizzle 16-elem (32B) chunks.
  for (int i = tid; i < FIN * HH; i += 256) {
    int k = i >> 7, n = i & 127;
    int c = (n >> 4) ^ (k & 7);
    w1s[(k << 7) + (c << 4) + (n & 15)] = to_bf16(W1[i]);
  }
  __syncthreads();

  const int wave = tid >> 5;
  const int lane = tid & 31;
  const long tile = (long)blockIdx.x * 8 + wave;
  if (tile >= MTILES) return;                 // wave-uniform, EXEC stays full

  const int half = lane >> 4;                 // K-halves for A fragment
  const int mrow = lane & 15;                 // A: M = lane % 16
  const float* arow = x + (tile * 16 + mrow) * (long)FIN;

  v8f acc[8];
#pragma unroll
  for (int nt = 0; nt < 8; ++nt) acc[nt] = (v8f)(0.0f);

  const int bsw = lane & 7;                   // B swizzle key: (k0+lane)&7

#pragma unroll
  for (int kc = 0; kc < 8; ++kc) {
    const int k0 = kc * 32;
    // ---- A fragment: lane holds row mrow, K = k0 + half*8 + {0..7, 16..23}
    const int kA = k0 + half * 8;
    float4 f0 = *(const float4*)(arow + kA);
    float4 f1 = *(const float4*)(arow + kA + 4);
    float4 f2 = *(const float4*)(arow + kA + 16);
    float4 f3 = *(const float4*)(arow + kA + 20);
    v16bf a;
    a[0]  = to_bf16(f0.x); a[1]  = to_bf16(f0.y); a[2]  = to_bf16(f0.z); a[3]  = to_bf16(f0.w);
    a[4]  = to_bf16(f1.x); a[5]  = to_bf16(f1.y); a[6]  = to_bf16(f1.z); a[7]  = to_bf16(f1.w);
    a[8]  = to_bf16(f2.x); a[9]  = to_bf16(f2.y); a[10] = to_bf16(f2.z); a[11] = to_bf16(f2.w);
    a[12] = to_bf16(f3.x); a[13] = to_bf16(f3.y); a[14] = to_bf16(f3.z); a[15] = to_bf16(f3.w);

    const __bf16* brow = &w1s[(k0 + half * 8) << 7];   // B: N = lane&15, K = k0 + half*8 + {0..7, 16..23}
#pragma unroll
    for (int nt = 0; nt < 8; ++nt) {
      v16bf b;
#pragma unroll
      for (int j = 0; j < 8; ++j) {
        const int off = (((nt ^ j) & 7) << 4) + (lane & 15);
        b[j] = brow[(j << 7) + off];
        b[j + 8] = brow[((j + 16) << 7) + off];
      }
      acc[nt] = __builtin_amdgcn_wmma_f32_16x16x32_bf16(
          false, a, false, b, (short)0, acc[nt], false, false);
    }
  }

  // ---- store: lane -> n = lane&15, rows r + 8*half
  const int ncol = lane & 15;
  float* orow = h1 + (tile * 16 + 8 * half) * (long)HH + ncol;
#pragma unroll
  for (int nt = 0; nt < 8; ++nt)
#pragma unroll
    for (int r = 0; r < 8; ++r)
      orow[(long)r * HH + nt * 16] = acc[nt][r];
}

// ---------------------------------------------------------------------------
// 3) wave-per-edge aggregation of 128 features: gather h1[src]*norm, atomic
//    scatter into ag1[dst]. L2-resident (h1 = 51MB << 192MB L2).
// ---------------------------------------------------------------------------
__global__ __launch_bounds__(256) void agg1_kernel(const float* __restrict__ h1,
                                                   const int* __restrict__ src,
                                                   const int* __restrict__ dst,
                                                   const float* __restrict__ dis,
                                                   float* __restrict__ ag1) {
  long g = (long)blockIdx.x * blockDim.x + threadIdx.x;
  long e = g >> 5;
  if (e >= EE) return;
  int lane = (int)(g & 31);
  int s = src[e], d = dst[e];
  float nrm = dis[s] * dis[d];
  float4 v = ((const float4*)(h1 + (long)s * HH))[lane];
  float* op = ag1 + (long)d * HH + lane * 4;
  atomicAdd(op + 0, v.x * nrm);
  atomicAdd(op + 1, v.y * nrm);
  atomicAdd(op + 2, v.z * nrm);
  atomicAdd(op + 3, v.w * nrm);
}

// ---------------------------------------------------------------------------
// 4) add self-loop term (dis[n]^2 * h1), bias, relu  (in place into ag1)
// ---------------------------------------------------------------------------
__global__ void bias_relu(const float* __restrict__ h1, const float* __restrict__ dis,
                          const float* __restrict__ b1, float* __restrict__ ag1) {
  long i = (long)blockIdx.x * blockDim.x + threadIdx.x;
  if (i >= (long)NN * HH) return;
  long n = i >> 7;
  int  j = (int)(i & 127);
  float dn = dis[n];
  float v = ag1[i] + h1[i] * dn * dn + b1[j];
  ag1[i] = v > 0.0f ? v : 0.0f;
}

// ---------------------------------------------------------------------------
// 5) h2 = relu_h @ W2  (128 -> 2): wave per node, shuffle reduction
// ---------------------------------------------------------------------------
__global__ __launch_bounds__(256) void gemm2_kernel(const float* __restrict__ hr,
                                                    const float* __restrict__ W2,
                                                    float* __restrict__ h2) {
  long g = (long)blockIdx.x * blockDim.x + threadIdx.x;
  long n = g >> 5;
  if (n >= NN) return;
  int lane = (int)(g & 31);
  float4 v = ((const float4*)(hr + n * HH))[lane];
  int k = lane * 4;
  float s0 = v.x * W2[(k + 0) * 2 + 0] + v.y * W2[(k + 1) * 2 + 0] +
             v.z * W2[(k + 2) * 2 + 0] + v.w * W2[(k + 3) * 2 + 0];
  float s1 = v.x * W2[(k + 0) * 2 + 1] + v.y * W2[(k + 1) * 2 + 1] +
             v.z * W2[(k + 2) * 2 + 1] + v.w * W2[(k + 3) * 2 + 1];
#pragma unroll
  for (int off = 16; off >= 1; off >>= 1) {
    s0 += __shfl_xor(s0, off, 32);
    s1 += __shfl_xor(s1, off, 32);
  }
  if (lane == 0) {
    h2[n * 2 + 0] = s0;
    h2[n * 2 + 1] = s1;
  }
}

// ---------------------------------------------------------------------------
// 6) thread-per-edge aggregation of 2 features
// ---------------------------------------------------------------------------
__global__ void agg2_kernel(const float* __restrict__ h2, const int* __restrict__ src,
                            const int* __restrict__ dst, const float* __restrict__ dis,
                            float* __restrict__ ag2) {
  long e = (long)blockIdx.x * blockDim.x + threadIdx.x;
  if (e >= EE) return;
  int s = src[e], d = dst[e];
  float nrm = dis[s] * dis[d];
  atomicAdd(&ag2[(long)d * 2 + 0], h2[(long)s * 2 + 0] * nrm);
  atomicAdd(&ag2[(long)d * 2 + 1], h2[(long)s * 2 + 1] * nrm);
}

// ---------------------------------------------------------------------------
// 7) self loop + b2, then mean-pool accumulation by graph id
// ---------------------------------------------------------------------------
__global__ void pool_kernel(const float* __restrict__ ag2, const float* __restrict__ h2,
                            const float* __restrict__ dis, const float* __restrict__ b2,
                            const int* __restrict__ batch,
                            float* __restrict__ gsum, float* __restrict__ gcnt) {
  long n = (long)blockIdx.x * blockDim.x + threadIdx.x;
  if (n >= NN) return;
  float dn = dis[n], d2 = dn * dn;
  float c0 = ag2[n * 2 + 0] + h2[n * 2 + 0] * d2 + b2[0];
  float c1 = ag2[n * 2 + 1] + h2[n * 2 + 1] * d2 + b2[1];
  int b = batch[n];
  atomicAdd(&gsum[b * 2 + 0], c0);
  atomicAdd(&gsum[b * 2 + 1], c1);
  atomicAdd(&gcnt[b], 1.0f);
}

// ---------------------------------------------------------------------------
// 8) mean + 2-class log_softmax
// ---------------------------------------------------------------------------
__global__ void final_kernel(const float* __restrict__ gsum, const float* __restrict__ gcnt,
                             float* __restrict__ out) {
  int g = blockIdx.x * blockDim.x + threadIdx.x;
  if (g >= GG) return;
  float c = fmaxf(gcnt[g], 1.0f);
  float p0 = gsum[g * 2 + 0] / c;
  float p1 = gsum[g * 2 + 1] / c;
  float m = fmaxf(p0, p1);
  float lse = m + logf(expf(p0 - m) + expf(p1 - m));
  out[g * 2 + 0] = p0 - lse;
  out[g * 2 + 1] = p1 - lse;
}

// ---------------------------------------------------------------------------
extern "C" void kernel_launch(void* const* d_in, const int* in_sizes, int n_in,
                              void* d_out, int out_size, void* d_ws, size_t ws_size,
                              hipStream_t stream) {
  (void)in_sizes; (void)n_in; (void)out_size; (void)ws_size;
  const float* x  = (const float*)d_in[0];
  const float* W1 = (const float*)d_in[1];
  const float* b1 = (const float*)d_in[2];
  const float* W2 = (const float*)d_in[3];
  const float* b2 = (const float*)d_in[4];
  const int*   ei = (const int*)d_in[5];
  const int*   batch = (const int*)d_in[6];
  const int* src = ei;
  const int* dst = ei + EE;
  float* out = (float*)d_out;

  // workspace carve-up (~105 MB)
  float* ws   = (float*)d_ws;
  float* h1   = ws;                            // N*128
  float* ag1  = h1  + (size_t)NN * HH;         // N*128
  float* h2   = ag1 + (size_t)NN * HH;         // N*2
  float* ag2  = h2  + (size_t)NN * CC;         // N*2
  float* deg  = ag2 + (size_t)NN * CC;         // N
  float* dis  = deg + NN;                      // N
  float* gsum = dis + NN;                      // G*2
  float* gcnt = gsum + (size_t)GG * CC;        // G

  const int T = 256;
  long elems = (long)NN * HH;

  init_ws<<<(unsigned)((elems + T - 1) / T), T, 0, stream>>>(ag1, ag2, deg, gsum, gcnt);
  deg_kernel<<<(EE + T - 1) / T, T, 0, stream>>>(dst, deg);
  dis_kernel<<<(NN + T - 1) / T, T, 0, stream>>>(deg, dis);
  gemm1_wmma<<<(MTILES + 7) / 8, T, 0, stream>>>(x, W1, h1);
  agg1_kernel<<<(unsigned)(((long)EE * 32 + T - 1) / T), T, 0, stream>>>(h1, src, dst, dis, ag1);
  bias_relu<<<(unsigned)((elems + T - 1) / T), T, 0, stream>>>(h1, dis, b1, ag1);
  gemm2_kernel<<<(unsigned)(((long)NN * 32 + T - 1) / T), T, 0, stream>>>(ag1, W2, h2);
  agg2_kernel<<<(EE + T - 1) / T, T, 0, stream>>>(h2, src, dst, dis, ag2);
  pool_kernel<<<(NN + T - 1) / T, T, 0, stream>>>(ag2, h2, dis, b2, batch, gsum, gcnt);
  final_kernel<<<1, GG, 0, stream>>>(gsum, gcnt, out);
}
